// GCGAT_v4pro_jit_49538152792982
// MI455X (gfx1250) — matrix operand order repaired and checked by reading, not compiled, not run on hardware
//
#include <hip/hip_runtime.h>
#include <hip/hip_bf16.h>
#include <math.h>

// ---------------------------------------------------------------------------
// Types / constants
// ---------------------------------------------------------------------------
typedef _Float16 h16;
typedef __attribute__((ext_vector_type(16))) _Float16 v16h;
typedef __attribute__((ext_vector_type(8)))  _Float16 v8h;
typedef __attribute__((ext_vector_type(8)))  float    v8f;

#define ACT_NONE     0
#define ACT_RELU     1
#define ACT_LEAKY01  2   // slope 0.01
#define ACT_LEAKY001 3   // slope 0.001

#define BM 128
#define BN 64
#define BK 32
#define LDSS 48          // LDS row stride in halfs (96B rows; 16B-aligned chunks)

static const int   NH    = 128;     // hidden H
static const int   BSZ   = 2048;    // num mols
static const long  NN    = 65536;   // data nodes
static const long  EE    = 131072;  // data edges
static const long  NFR   = 65536;   // frag nodes
static const long  EFR   = 131072;  // frag edges
static const long  NJT   = 8192;    // junction nodes / frag graphs
static const long  EJT   = 12288;   // junction edges
#define BN_INV 0.9999950000374997f  // 1/sqrt(1+1e-5)

// ---------------------------------------------------------------------------
// Staging helpers: branchless 8-float row-chunk load (clamped + masked),
// packed f16 store to LDS (single ds_store_b128).
// ---------------------------------------------------------------------------
__device__ __forceinline__ void load8(const float* __restrict__ P, int rowstride,
                                      int r, int rmax, int kbase, int K, int k_s,
                                      float v[8]) {
    const bool rok = r < rmax;
    const int  rc  = rok ? r : (rmax - 1);
    const float* p = P + (long)rc * rowstride + kbase;
    if (rok && (kbase + k_s + 8 <= K)) {
        // rows of every GEMM operand are 16B aligned; k_s is a multiple of 8 floats
        const float4 f0 = *(const float4*)(p + k_s);
        const float4 f1 = *(const float4*)(p + k_s + 4);
        v[0] = f0.x; v[1] = f0.y; v[2] = f0.z; v[3] = f0.w;
        v[4] = f1.x; v[5] = f1.y; v[6] = f1.z; v[7] = f1.w;
    } else {
        #pragma unroll
        for (int j = 0; j < 8; ++j) {
            const int k  = kbase + k_s + j;
            const int kc = (k < K) ? (k_s + j) : 0;   // clamped, always in-range
            const float t = p[kc];                     // unconditional load
            v[j] = (rok && k < K) ? t : 0.0f;          // mask via cndmask
        }
    }
}

__device__ __forceinline__ void store8(h16* __restrict__ base, int row_s, int k_s,
                                       const float v[8]) {
    v8h h;
    #pragma unroll
    for (int j = 0; j < 8; ++j) h[j] = (h16)v[j];
    *(v8h*)(&base[row_s * LDSS + k_s]) = h;            // ds_store_b128
}

// fragment loaders (layouts per CDNA5 ISA 7.12.2)
__device__ __forceinline__ v16h frag_a(const h16* __restrict__ buf, int row, int hi) {
    // 16x32 f16 A: lane<16 holds K {0..7,16..23}; hi lanes hold K {8..15,24..31}
    const h16* base = &buf[row * LDSS + hi * 8];
    v8h lo  = *(const v8h*)(base);
    v8h hi8 = *(const v8h*)(base + 16);
    v16h a;
    #pragma unroll
    for (int j = 0; j < 8; ++j) { a[j] = lo[j]; a[j + 8] = hi8[j]; }
    return a;
}
__device__ __forceinline__ v16h frag_b(const h16* __restrict__ buf, int col, int hi) {
    // 32x16 f16 B: lane<16 holds K {0..7,16..23}; hi lanes hold K {8..15,24..31}
    const h16* p0 = &buf[col * LDSS + hi * 8];
    v8h x0 = *(const v8h*)(p0);
    v8h x1 = *(const v8h*)(p0 + 16);
    v16h b;
    #pragma unroll
    for (int j = 0; j < 8; ++j) { b[j] = x0[j]; b[j + 8] = x1[j]; }
    return b;
}

// ---------------------------------------------------------------------------
// WMMA GEMM:  C[M,N] = act( (A[M,K] @ W[N,K]^T + bias) * bn_scale + bn_shift )
// A row-major (lda), W row-major [N,K] with row stride ldw, C row-major (ldc).
// f32 in/out, f16 WMMA core with f32 accumulate.
// Block tile 128x64, 8 waves of 32x32 (4 WMMA each), double-buffered LDS.
// ---------------------------------------------------------------------------
__global__ __launch_bounds__(256)
void gemm_wmma_kernel(const float* __restrict__ A, int lda,
                      const float* __restrict__ Wt, int ldw,
                      const float* __restrict__ bias,
                      const float* __restrict__ bng, const float* __restrict__ bnb,
                      float* __restrict__ C, int ldc,
                      int M, int K, int Nout, int act)
{
    __shared__ h16 As[2][BM * LDSS];
    __shared__ h16 Bs[2][BN * LDSS];

    const int tid  = threadIdx.x;
    const int wave = tid >> 5;
    const int lane = tid & 31;
    const int hl   = lane & 15;     // column / row within lane group
    const int hi   = lane >> 4;     // lane group
    const int bm   = blockIdx.x * BM;
    const int bnc  = blockIdx.y * BN;
    const int wm   = wave & 3;      // 4 wave-rows of 32
    const int wn   = wave >> 2;     // 2 wave-cols of 32

    v8f acc00 = {}, acc01 = {}, acc10 = {}, acc11 = {};

    const int row_s = tid >> 2;        // 0..63 staged row
    const int k_s   = (tid & 3) * 8;   // 0,8,16,24 staged k chunk
    const int grA0  = bm + row_s;          // staged A rows (two half-tiles)
    const int grA1  = bm + row_s + 64;
    const int gnW   = bnc + row_s;         // staged W row (output col)

    float va0[8], va1[8], vw[8];

    // prologue: stage tile 0
    load8(A, lda, grA0, M, 0, K, k_s, va0);
    load8(A, lda, grA1, M, 0, K, k_s, va1);
    load8(Wt, ldw, gnW, Nout, 0, K, k_s, vw);
    store8(&As[0][0],            row_s,      k_s, va0);
    store8(&As[0][0],            row_s + 64, k_s, va1);
    store8(&Bs[0][0],            row_s,      k_s, vw);
    __syncthreads();

    const int nk = (K + BK - 1) / BK;
    for (int t = 0; t < nk; ++t) {
        const int cur = t & 1;

        // prefetch tile t+1 into registers (overlaps the WMMAs below)
        if (t + 1 < nk) {
            load8(A, lda, grA0, M, (t + 1) * BK, K, k_s, va0);
            load8(A, lda, grA1, M, (t + 1) * BK, K, k_s, va1);
            load8(Wt, ldw, gnW, Nout, (t + 1) * BK, K, k_s, vw);
        }

        // fragments for this wave's 32x32 tile
        const v16h a0 = frag_a(&As[cur][0], wm * 32 + hl,      hi);
        const v16h a1 = frag_a(&As[cur][0], wm * 32 + 16 + hl, hi);
        const v16h b0 = frag_b(&Bs[cur][0], wn * 32 + hl,      hi);
        const v16h b1 = frag_b(&Bs[cur][0], wn * 32 + 16 + hl, hi);

        acc00 = __builtin_amdgcn_wmma_f32_16x16x32_f16(false, a0, false, b0, (short)0, acc00, false, false);
        acc01 = __builtin_amdgcn_wmma_f32_16x16x32_f16(false, a0, false, b1, (short)0, acc01, false, false);
        acc10 = __builtin_amdgcn_wmma_f32_16x16x32_f16(false, a1, false, b0, (short)0, acc10, false, false);
        acc11 = __builtin_amdgcn_wmma_f32_16x16x32_f16(false, a1, false, b1, (short)0, acc11, false, false);

        // store prefetched tile into the other LDS buffer
        if (t + 1 < nk) {
            store8(&As[cur ^ 1][0], row_s,      k_s, va0);
            store8(&As[cur ^ 1][0], row_s + 64, k_s, va1);
            store8(&Bs[cur ^ 1][0], row_s,      k_s, vw);
        }
        __syncthreads();
    }

    // ---- epilogue: C layout lane(0..15)=N col, VGPR v = M row (hi lanes M+8) ----
    #pragma unroll
    for (int rb = 0; rb < 2; ++rb) {
        #pragma unroll
        for (int half = 0; half < 2; ++half) {
            v8f acc = rb ? (half ? acc11 : acc10) : (half ? acc01 : acc00);
            const int gc = bnc + wn * 32 + half * 16 + hl;
            if (gc < Nout) {
                float bv = bias ? bias[gc] : 0.0f;
                float sc = 1.0f, sh = 0.0f;
                if (bng) { sc = bng[gc] * BN_INV; sh = bnb[gc]; }
                #pragma unroll
                for (int v = 0; v < 8; ++v) {
                    const int grr = bm + wm * 32 + rb * 16 + hi * 8 + v;
                    if (grr < M) {
                        float t = acc[v] + bv;
                        t = t * sc + sh;
                        if (act == ACT_RELU)          t = t > 0.0f ? t : 0.0f;
                        else if (act == ACT_LEAKY01)  t = t >= 0.0f ? t : 0.01f * t;
                        else if (act == ACT_LEAKY001) t = t >= 0.0f ? t : 0.001f * t;
                        C[(long)grr * ldc + gc] = t;
                    }
                }
            }
        }
    }
}

// ---------------------------------------------------------------------------
// Elementwise / graph kernels
// ---------------------------------------------------------------------------
__device__ __forceinline__ void atomicMaxF(float* addr, float val) {
    if (val >= 0.0f) atomicMax((int*)addr, __float_as_int(val));
    else             atomicMin((unsigned int*)addr, __float_as_uint(val));
}

__global__ void fill_kernel(float* __restrict__ p, float v, long n) {
    long t = (long)blockIdx.x * blockDim.x + threadIdx.x;
    if (t < n) p[t] = v;
}

__global__ void fill_rows_kernel(float* __restrict__ dst, int ldd,
                                 const float* __restrict__ bias, int rows, int cols) {
    long t = (long)blockIdx.x * blockDim.x + threadIdx.x;
    long tot = (long)rows * cols;
    if (t >= tot) return;
    int r = (int)(t / cols), c = (int)(t % cols);
    dst[(long)r * ldd + c] = bias ? bias[c] : 0.0f;
}

// dst[r,c] = src[idx[r], c]   (idx==null -> identity copy with stride change)
__global__ void gather_rows_kernel(float* __restrict__ dst, int ldd,
                                   const int* __restrict__ idx,
                                   const float* __restrict__ src, int lds,
                                   int rows, int cols) {
    long t = (long)blockIdx.x * blockDim.x + threadIdx.x;
    long tot = (long)rows * cols;
    if (t >= tot) return;
    int r = (int)(t / cols), c = (int)(t % cols);
    int s = idx ? idx[r] : r;
    dst[(long)r * ldd + c] = src[(long)s * lds + c];
}

// out[seg[r], c] += x[r, c]
__global__ void seg_add_kernel(float* __restrict__ out, int ldo,
                               const float* __restrict__ x, int ldx,
                               const int* __restrict__ seg, int rows, int cols) {
    long t = (long)blockIdx.x * blockDim.x + threadIdx.x;
    long tot = (long)rows * cols;
    if (t >= tot) return;
    int r = (int)(t / cols), c = (int)(t % cols);
    atomicAdd(&out[(long)seg[r] * ldo + c], x[(long)r * ldx + c]);
}

// out[dst[e], c] += msg[midx?:e, c] * w[e]
__global__ void scatter_wadd_kernel(float* __restrict__ out, int ldo,
                                    const float* __restrict__ msg, int ldm,
                                    const int* __restrict__ midx,
                                    const float* __restrict__ w,
                                    const int* __restrict__ dst, int E, int cols) {
    long t = (long)blockIdx.x * blockDim.x + threadIdx.x;
    long tot = (long)E * cols;
    if (t >= tot) return;
    int e = (int)(t / cols), c = (int)(t % cols);
    int mi = midx ? midx[e] : e;
    atomicAdd(&out[(long)dst[e] * ldo + c], msg[(long)mi * ldm + c] * w[e]);
}

// wave-per-row matvec: out[r] = dot(X[r,:], v)
__global__ void matvec_kernel(const float* __restrict__ X, int ld,
                              const float* __restrict__ v,
                              float* __restrict__ out, int rows, int K) {
    int w = (int)(((long)blockIdx.x * blockDim.x + threadIdx.x) >> 5);
    int lane = threadIdx.x & 31;
    if (w >= rows) return;
    const float* xp = X + (long)w * ld;
    float s = 0.0f;
    for (int k = lane; k < K; k += 32) s += xp[k] * v[k];
    for (int off = 16; off > 0; off >>= 1) s += __shfl_xor(s, off, 32);
    if (lane == 0) out[w] = s;
}

// a[e] = leaky( dot(xj[e,:], att) + r[dst[e]], 0.01 )
__global__ void edge_logit_dot_kernel(const float* __restrict__ xj,
                                      const float* __restrict__ att,
                                      const float* __restrict__ r,
                                      const int* __restrict__ dst,
                                      float* __restrict__ a, int E, int K) {
    int e = (int)(((long)blockIdx.x * blockDim.x + threadIdx.x) >> 5);
    int lane = threadIdx.x & 31;
    if (e >= E) return;
    const float* xp = xj + (long)e * K;
    float s = 0.0f;
    for (int k = lane; k < K; k += 32) s += xp[k] * att[k];
    for (int off = 16; off > 0; off >>= 1) s += __shfl_xor(s, off, 32);
    if (lane == 0) {
        float v = s + r[dst[e]];
        a[e] = v >= 0.0f ? v : 0.01f * v;
    }
}

// a[e] = leaky( as[src?:e] + ad[dst[e]], 0.01 )
__global__ void edge_logit_pair_kernel(const float* __restrict__ as_,
                                       const float* __restrict__ ad_,
                                       const int* __restrict__ src,
                                       const int* __restrict__ dst,
                                       float* __restrict__ a, int E) {
    int e = (int)((long)blockIdx.x * blockDim.x + threadIdx.x);
    if (e >= E) return;
    int si = src ? src[e] : e;
    float v = as_[si] + ad_[dst[e]];
    a[e] = v >= 0.0f ? v : 0.01f * v;
}

__global__ void seg_max_kernel(const float* __restrict__ a, const int* __restrict__ seg,
                               float* __restrict__ m, int E) {
    int e = (int)((long)blockIdx.x * blockDim.x + threadIdx.x);
    if (e >= E) return;
    atomicMaxF(&m[seg[e]], a[e]);
}

__global__ void seg_expsum_kernel(float* __restrict__ a, const int* __restrict__ seg,
                                  const float* __restrict__ m, float* __restrict__ s, int E) {
    int e = (int)((long)blockIdx.x * blockDim.x + threadIdx.x);
    if (e >= E) return;
    float x = expf(a[e] - m[seg[e]]);
    a[e] = x;
    atomicAdd(&s[seg[e]], x);
}

__global__ void seg_norm_kernel(float* __restrict__ a, const int* __restrict__ seg,
                                const float* __restrict__ s, int E) {
    int e = (int)((long)blockIdx.x * blockDim.x + threadIdx.x);
    if (e >= E) return;
    a[e] = a[e] / (s[seg[e]] + 1e-16f);
}

// GRU pointwise from gi=[r|z|n] (x-path) and gh (h-path), hidden in/out [n,128]
__global__ void gru_pointwise_kernel(const float* __restrict__ gi,
                                     const float* __restrict__ gh,
                                     const float* __restrict__ hprev,
                                     float* __restrict__ hnew,
                                     int n, int relu_out) {
    long t = (long)blockIdx.x * blockDim.x + threadIdx.x;
    long tot = (long)n * NH;
    if (t >= tot) return;
    int r = (int)(t >> 7), c = (int)(t & 127);
    const float* gir = gi + (long)r * 384;
    const float* ghr = gh + (long)r * 384;
    float rg = 1.0f / (1.0f + expf(-(gir[c]       + ghr[c])));
    float zg = 1.0f / (1.0f + expf(-(gir[c + 128] + ghr[c + 128])));
    float ng = tanhf(gir[c + 256] + rg * ghr[c + 256]);
    float h = (1.0f - zg) * ng + zg * hprev[(long)r * NH + c];
    if (relu_out) h = h > 0.0f ? h : 0.0f;
    hnew[(long)r * NH + c] = h;
}

__global__ void act_kernel(float* __restrict__ x, long n, int mode) { // 0=relu 1=elu
    long t = (long)blockIdx.x * blockDim.x + threadIdx.x;
    if (t >= n) return;
    float v = x[t];
    x[t] = (mode == 0) ? (v > 0.0f ? v : 0.0f)
                       : (v > 0.0f ? v : expm1f(v));
}

// in-place eval BatchNorm (applied after activation, for pred1)
__global__ void bn_kernel(float* __restrict__ x, const float* __restrict__ g,
                          const float* __restrict__ b, int rows, int cols) {
    long t = (long)blockIdx.x * blockDim.x + threadIdx.x;
    long tot = (long)rows * cols;
    if (t >= tot) return;
    int c = (int)(t % cols);
    x[t] = x[t] * (g[c] * BN_INV) + b[c];
}

__global__ void mean2_relu_kernel(const float* __restrict__ a, const float* __restrict__ b,
                                  float* __restrict__ dst, int ldd, int rows, int cols) {
    long t = (long)blockIdx.x * blockDim.x + threadIdx.x;
    long tot = (long)rows * cols;
    if (t >= tot) return;
    int r = (int)(t / cols), c = (int)(t % cols);
    float v = 0.5f * (a[(long)r * cols + c] + b[(long)r * cols + c]);
    dst[(long)r * ldd + c] = v > 0.0f ? v : 0.0f;
}

// ---------------------------------------------------------------------------
// Host-side parameter views (flattened pytree walk, make_params insertion order)
// ---------------------------------------------------------------------------
struct Lin  { const float *w, *b; };
struct Bnp  { const float *g, *b; };
struct Gru  { const float *w_ih, *w_hh, *b_ih, *b_hh; };
struct Gat  { const float *w, *att_src, *att_dst, *bias; };
struct Gate { const float *lin1_w, *att_l, *att_r, *lin2_w, *bias; };
struct AfpP {
    Lin lin1; Gate gate; Gru gru0;
    Gat aconv[2]; Gru agru[2]; int n_gat;
    Gat mol_conv; Gru mol_gru; Lin lin2;
};
struct Walker {
    void* const* din; int i;
    const float* f() { return (const float*)din[i++]; }
};
static Lin  rdLin(Walker& p)  { Lin l;  l.w = p.f(); l.b = p.f(); return l; }
static Bnp  rdBn(Walker& p)   { Bnp b;  b.g = p.f(); b.b = p.f(); return b; }
static Gru  rdGru(Walker& p)  { Gru g;  g.w_ih = p.f(); g.w_hh = p.f(); g.b_ih = p.f(); g.b_hh = p.f(); return g; }
static Gat  rdGat(Walker& p)  { Gat g;  g.w = p.f(); g.att_src = p.f(); g.att_dst = p.f(); g.bias = p.f(); return g; }
static Gate rdGate(Walker& p) { Gate g; g.lin1_w = p.f(); g.att_l = p.f(); g.att_r = p.f(); g.lin2_w = p.f(); g.bias = p.f(); return g; }
static AfpP rdAfp(Walker& p, int n_atom) {
    AfpP a; a.lin1 = rdLin(p); a.gate = rdGate(p); a.gru0 = rdGru(p);
    a.n_gat = n_atom - 1;
    for (int k = 0; k < a.n_gat; ++k) a.aconv[k] = rdGat(p);
    for (int k = 0; k < a.n_gat; ++k) a.agru[k]  = rdGru(p);
    a.mol_conv = rdGat(p); a.mol_gru = rdGru(p); a.lin2 = rdLin(p);
    return a;
}

// ---------------------------------------------------------------------------
// Workspace
// ---------------------------------------------------------------------------
struct WS {
    float *bufA, *bufB;        // E*256 and E*128, contiguous (GRU gates alias them)
    float *giN, *ghN;          // aliases into bufA/bufB region: N*384 each
    float *nx, *nh;            // N*128
    float *rbuf, *rbuf2;       // N
    float *abuf;               // E
    float *mbuf, *sbuf;        // N (segment max/sum)
    float *hg, *outg;          // Gmax*128
    float *gig, *ghg;          // Gmax*384
    float *xn, *ean;           // channel node/edge transforms (reused as xcat/eaejt)
    float *cat_o, *cat_f;      // head concat buffers
    float *graph_frag;         // NJT*128
    float *xpbuf;              // NJT*128
    float *jtouts;             // 2*B*128
    float *cat3;               // B*384
    float *dbuf, *dbuf2;       // B*192
    float *jxpad;              // NJT*220 (junction_x, padded for 16B-aligned rows)
    float *fwpad;              // 128*220 (jt frag_lin weight, padded)
};

// ---------------------------------------------------------------------------
// Host launch helpers
// ---------------------------------------------------------------------------
static inline dim3 g1(long n) { return dim3((unsigned)((n + 255) / 256)); }

static void gemm(hipStream_t st, const float* A, int lda, const float* Wt, int ldw,
                 const float* bias, float* C, int ldc,
                 int M, int K, int Nout, int act,
                 const float* bng = nullptr, const float* bnb = nullptr) {
    dim3 grid((M + BM - 1) / BM, (Nout + BN - 1) / BN);
    gemm_wmma_kernel<<<grid, dim3(256), 0, st>>>(A, lda, Wt, ldw, bias, bng, bnb,
                                                 C, ldc, M, K, Nout, act);
}

static void seg_softmax_h(hipStream_t st, float* a, const int* seg, int E,
                          float* m, float* s, int nseg) {
    fill_kernel<<<g1(nseg), 256, 0, st>>>(m, -3.402823466e38f, nseg);
    seg_max_kernel<<<g1(E), 256, 0, st>>>(a, seg, m, E);
    fill_kernel<<<g1(nseg), 256, 0, st>>>(s, 0.0f, nseg);
    seg_expsum_kernel<<<g1(E), 256, 0, st>>>(a, seg, m, s, E);
    seg_norm_kernel<<<g1(E), 256, 0, st>>>(a, seg, s, E);
}

static void run_gru(hipStream_t st, const Gru& G, const float* xin, float* hid,
                    float* gi, float* gh, int n, int relu) {
    gemm(st, xin, NH, G.w_ih, NH, G.b_ih, gi, 384, n, NH, 384, ACT_NONE);
    gemm(st, hid, NH, G.w_hh, NH, G.b_hh, gh, 384, n, NH, 384, ACT_NONE);
    gru_pointwise_kernel<<<g1((long)n * NH), 256, 0, st>>>(gi, gh, hid, hid, n, relu);
}

// One full AttentiveFP head (eval mode).
static void run_afp(hipStream_t st, const AfpP& P, WS& W,
                    const float* x_in, int n, int in_dim,
                    const int* src, const int* dst, int E,
                    const float* ea, int edim,
                    const int* batch, int g, int T,
                    float* out_c, int ldc_out) {
    // x = leaky(lin1(x_in))
    gemm(st, x_in, in_dim, P.lin1.w, in_dim, P.lin1.b, W.nx, NH, n, in_dim, NH, ACT_LEAKY01);

    // ---- GATE conv ----
    const int K1 = NH + edim;
    gather_rows_kernel<<<g1((long)E * NH), 256, 0, st>>>(W.bufA, K1, src, W.nx, NH, E, NH);
    gather_rows_kernel<<<g1((long)E * edim), 256, 0, st>>>(W.bufA + NH, K1, nullptr, ea, edim, E, edim);
    gemm(st, W.bufA, K1, P.gate.lin1_w, K1, nullptr, W.bufB, NH, E, K1, NH, ACT_LEAKY01);      // xj
    matvec_kernel<<<g1((long)n * 32), 256, 0, st>>>(W.nx, NH, P.gate.att_r, W.rbuf, n, NH);    // r
    edge_logit_dot_kernel<<<g1((long)E * 32), 256, 0, st>>>(W.bufB, P.gate.att_l, W.rbuf, dst, W.abuf, E, NH);
    seg_softmax_h(st, W.abuf, dst, E, W.mbuf, W.sbuf, n);
    gemm(st, W.bufB, NH, P.gate.lin2_w, NH, nullptr, W.bufA, NH, E, NH, NH, ACT_NONE);         // msg (bufA reuse)
    fill_rows_kernel<<<g1((long)n * NH), 256, 0, st>>>(W.nh, NH, P.gate.bias, n, NH);
    scatter_wadd_kernel<<<g1((long)E * NH), 256, 0, st>>>(W.nh, NH, W.bufA, NH, nullptr, W.abuf, dst, E, NH);
    act_kernel<<<g1((long)n * NH), 256, 0, st>>>(W.nh, (long)n * NH, 1);                       // elu
    run_gru(st, P.gru0, W.nh, W.nx, W.giN, W.ghN, n, 1);                                       // x=relu(gru)

    // ---- atom GAT layers ----
    for (int l = 0; l < P.n_gat; ++l) {
        const Gat& Cv = P.aconv[l];
        gemm(st, W.nx, NH, Cv.w, NH, nullptr, W.bufB, NH, n, NH, NH, ACT_NONE);                // hs
        matvec_kernel<<<g1((long)n * 32), 256, 0, st>>>(W.bufB, NH, Cv.att_src, W.rbuf, n, NH);
        matvec_kernel<<<g1((long)n * 32), 256, 0, st>>>(W.bufB, NH, Cv.att_dst, W.rbuf2, n, NH);
        edge_logit_pair_kernel<<<g1(E), 256, 0, st>>>(W.rbuf, W.rbuf2, src, dst, W.abuf, E);
        seg_softmax_h(st, W.abuf, dst, E, W.mbuf, W.sbuf, n);
        fill_rows_kernel<<<g1((long)n * NH), 256, 0, st>>>(W.nh, NH, Cv.bias, n, NH);
        scatter_wadd_kernel<<<g1((long)E * NH), 256, 0, st>>>(W.nh, NH, W.bufB, NH, src, W.abuf, dst, E, NH);
        act_kernel<<<g1((long)n * NH), 256, 0, st>>>(W.nh, (long)n * NH, 1);
        run_gru(st, P.agru[l], W.nh, W.nx, W.giN, W.ghN, n, 1);
    }

    // ---- graph pooling ----
    fill_rows_kernel<<<g1((long)g * NH), 256, 0, st>>>(W.outg, NH, nullptr, g, NH);
    seg_add_kernel<<<g1((long)n * NH), 256, 0, st>>>(W.outg, NH, W.nx, NH, batch, n, NH);
    act_kernel<<<g1((long)g * NH), 256, 0, st>>>(W.outg, (long)g * NH, 0);                     // relu

    // ---- molecule readout timesteps ----
    for (int t = 0; t < T; ++t) {
        gemm(st, W.nx, NH, P.mol_conv.w, NH, nullptr, W.bufB, NH, n, NH, NH, ACT_NONE);        // hs [n]
        gemm(st, W.outg, NH, P.mol_conv.w, NH, nullptr, W.hg, NH, g, NH, NH, ACT_NONE);        // hd [g]
        matvec_kernel<<<g1((long)n * 32), 256, 0, st>>>(W.bufB, NH, P.mol_conv.att_src, W.rbuf, n, NH);
        matvec_kernel<<<g1((long)g * 32), 256, 0, st>>>(W.hg, NH, P.mol_conv.att_dst, W.rbuf2, g, NH);
        edge_logit_pair_kernel<<<g1(n), 256, 0, st>>>(W.rbuf, W.rbuf2, nullptr, batch, W.abuf, n);
        seg_softmax_h(st, W.abuf, batch, n, W.mbuf, W.sbuf, g);
        fill_rows_kernel<<<g1((long)g * NH), 256, 0, st>>>(W.hg, NH, P.mol_conv.bias, g, NH);
        scatter_wadd_kernel<<<g1((long)n * NH), 256, 0, st>>>(W.hg, NH, W.bufB, NH, nullptr, W.abuf, batch, n, NH);
        act_kernel<<<g1((long)g * NH), 256, 0, st>>>(W.hg, (long)g * NH, 1);                   // elu
        run_gru(st, P.mol_gru, W.hg, W.outg, W.gig, W.ghg, g, 1);                              // out=relu(gru)
    }

    // ---- final per-graph projection ----
    gemm(st, W.outg, NH, P.lin2.w, NH, P.lin2.b, out_c, ldc_out, g, NH, NH, ACT_NONE);
}

// ---------------------------------------------------------------------------
// kernel_launch — full GCGAT forward
// ---------------------------------------------------------------------------
extern "C" void kernel_launch(void* const* d_in, const int* in_sizes, int n_in,
                              void* d_out, int out_size, void* d_ws, size_t ws_size,
                              hipStream_t stream) {
    (void)in_sizes; (void)n_in; (void)out_size; (void)ws_size;
    hipStream_t st = stream;

    // ---- inputs ----
    const float* data_x   = (const float*)d_in[0];
    const int*   dei      = (const int*)d_in[1];
    const float* data_ea  = (const float*)d_in[2];
    const int*   dbatch   = (const int*)d_in[3];
    const float* frag_x   = (const float*)d_in[4];
    const int*   fei      = (const int*)d_in[5];
    const float* frag_ea  = (const float*)d_in[6];
    const int*   fbatch   = (const int*)d_in[7];
    const float* junc_x   = (const float*)d_in[8];
    const int*   jei      = (const int*)d_in[9];
    const float* junc_ea  = (const float*)d_in[10];
    const int*   jbatch   = (const int*)d_in[11];
    // d_in[12]=motif_nodes (unused), d_in[13]=global_feats (unused),
    // d_in[14]=num_mols, d_in[15]=num_frag_graphs (compile-time constants)

    const int *dsrc = dei, *ddst = dei + EE;
    const int *fsrc = fei, *fdst = fei + EFR;
    const int *jsrc = jei, *jdst = jei + EJT;

    // ---- params (flattened pytree, insertion order) ----
    Walker P{d_in, 16};
    // origin
    Lin o_node_lin = rdLin(P); Bnp o_node_bn = rdBn(P);
    Lin o_edge_lin = rdLin(P); Bnp o_edge_bn = rdBn(P);
    AfpP o_heads[2]; o_heads[0] = rdAfp(P, 3); o_heads[1] = rdAfp(P, 3);
    Lin o_attend = rdLin(P); Bnp o_attend_bn = rdBn(P);
    // frag
    AfpP f_heads[2]; f_heads[0] = rdAfp(P, 2); f_heads[1] = rdAfp(P, 2);
    Lin f_attend = rdLin(P); Bnp f_attend_bn = rdBn(P);
    // jt
    Lin j_frag_lin = rdLin(P); Bnp j_frag_bn = rdBn(P);
    Lin j_edge_lin = rdLin(P); Bnp j_edge_bn = rdBn(P);
    Lin j_proj[2]; AfpP j_afp[2];
    for (int h = 0; h < 2; ++h) { j_proj[h] = rdLin(P); j_afp[h] = rdAfp(P, 2); }
    // pred
    Lin p1_lin = rdLin(P); Bnp p1_bn = rdBn(P);
    Lin p2_mid = rdLin(P); Lin p2_out = rdLin(P);

    // ---- workspace carve ----
    float* base = (float*)d_ws;
    long off = 0;
    auto alloc = [&](long nfl) { float* p = base + off; off += nfl; return p; };
    WS W;
    W.bufA  = alloc(EE * 256);          // gather/concat & msg
    W.bufB  = alloc(EE * NH);           // xj / hs (contiguous after bufA)
    W.giN   = W.bufA;                   // GRU gates alias dead conv buffers
    W.ghN   = W.bufA + NN * 384;
    W.nx    = alloc(NN * NH);
    W.nh    = alloc(NN * NH);
    W.rbuf  = alloc(NN);
    W.rbuf2 = alloc(NN);
    W.abuf  = alloc(EE);
    W.mbuf  = alloc(NN);
    W.sbuf  = alloc(NN);
    W.hg    = alloc(NJT * NH);
    W.outg  = alloc(NJT * NH);
    W.gig   = alloc(NJT * 384);
    W.ghg   = alloc(NJT * 384);
    W.xn    = alloc(NN * NH);           // reused as jt xcat [NJT,256]
    W.ean   = alloc(EE * NH);           // reused as jt eae  [EJT,128]
    W.cat_o = alloc((long)BSZ * 256);
    W.cat_f = alloc(NJT * 256);
    W.graph_frag = alloc(NJT * NH);
    W.xpbuf = alloc(NJT * NH);
    W.jtouts = alloc(2L * BSZ * NH);
    W.cat3  = alloc((long)BSZ * 384);
    W.dbuf  = alloc((long)BSZ * 192);
    W.dbuf2 = alloc((long)BSZ * 192);
    W.jxpad = alloc(NJT * 220);
    W.fwpad = alloc(128L * 220);

    // ================= origin channel =================
    gemm(st, data_x, 44, o_node_lin.w, 44, o_node_lin.b, W.xn, NH, (int)NN, 44, NH,
         ACT_LEAKY01, o_node_bn.g, o_node_bn.b);
    gemm(st, data_ea, 12, o_edge_lin.w, 12, o_edge_lin.b, W.ean, NH, (int)EE, 12, NH,
         ACT_LEAKY01, o_edge_bn.g, o_edge_bn.b);
    for (int h = 0; h < 2; ++h)
        run_afp(st, o_heads[h], W, W.xn, (int)NN, NH, dsrc, ddst, (int)EE,
                W.ean, NH, dbatch, BSZ, 3, W.cat_o + h * NH, 256);
    gemm(st, W.cat_o, 256, o_attend.w, 256, o_attend.b, W.cat3, 384, BSZ, 256, NH,
         ACT_RELU, o_attend_bn.g, o_attend_bn.b);                     // cat3[:,0:128]

    // ================= frag channel =================
    for (int h = 0; h < 2; ++h)
        run_afp(st, f_heads[h], W, frag_x, (int)NFR, 44, fsrc, fdst, (int)EFR,
                frag_ea, 12, fbatch, (int)NJT, 2, W.cat_f + h * NH, 256);
    gemm(st, W.cat_f, 256, f_attend.w, 256, f_attend.b, W.graph_frag, NH, (int)NJT, 256, NH,
         ACT_RELU, f_attend_bn.g, f_attend_bn.b);

    // frag_res = segment_sum(graph_frag, junction_batch) -> cat3[:,128:256]
    fill_rows_kernel<<<g1((long)BSZ * NH), 256, 0, st>>>(W.cat3 + 128, 384, nullptr, BSZ, NH);
    seg_add_kernel<<<g1(NJT * NH), 256, 0, st>>>(W.cat3 + 128, 384, W.graph_frag, NH, jbatch, (int)NJT, NH);

    // ================= junction-tree channel =================
    float* xcat = W.xn;      // [NJT,256] = [graph_frag | mn]
    float* eae  = W.ean;     // [EJT,128]
    // pad 219-wide operands to 220 stride so every GEMM row is 16B-aligned
    gather_rows_kernel<<<g1(NJT * 219), 256, 0, st>>>(W.jxpad, 220, nullptr, junc_x, 219, (int)NJT, 219);
    gather_rows_kernel<<<g1(128L * 219), 256, 0, st>>>(W.fwpad, 220, nullptr, j_frag_lin.w, 219, 128, 219);
    gemm(st, W.jxpad, 220, W.fwpad, 220, j_frag_lin.b, xcat + 128, 256, (int)NJT, 219, NH,
         ACT_LEAKY01, j_frag_bn.g, j_frag_bn.b);                       // mn
    gather_rows_kernel<<<g1(NJT * NH), 256, 0, st>>>(xcat, 256, nullptr, W.graph_frag, NH, (int)NJT, NH);
    gemm(st, junc_ea, 12, j_edge_lin.w, 12, j_edge_lin.b, eae, NH, (int)EJT, 12, NH,
         ACT_LEAKY01, j_edge_bn.g, j_edge_bn.b);
    for (int h = 0; h < 2; ++h) {
        gemm(st, xcat, 256, j_proj[h].w, 256, j_proj[h].b, W.xpbuf, NH, (int)NJT, 256, NH, ACT_NONE);
        run_afp(st, j_afp[h], W, W.xpbuf, (int)NJT, NH, jsrc, jdst, (int)EJT,
                eae, NH, jbatch, BSZ, 2, W.jtouts + (long)h * BSZ * NH, NH);
    }
    mean2_relu_kernel<<<g1((long)BSZ * NH), 256, 0, st>>>(
        W.jtouts, W.jtouts + (long)BSZ * NH, W.cat3 + 256, 384, BSZ, NH); // cat3[:,256:384]

    // ================= prediction MLP =================
    gemm(st, W.cat3, 384, p1_lin.w, 384, p1_lin.b, W.dbuf, 192, BSZ, 384, 192, ACT_LEAKY001);
    bn_kernel<<<g1((long)BSZ * 192), 256, 0, st>>>(W.dbuf, p1_bn.g, p1_bn.b, BSZ, 192);
    gemm(st, W.dbuf, 192, p2_mid.w, 192, p2_mid.b, W.dbuf2, 192, BSZ, 192, 192, ACT_LEAKY001);
    gemm(st, W.dbuf2, 192, p2_out.w, 192, p2_out.b, (float*)d_out, 1, BSZ, 192, 1, ACT_NONE);
}
